// MetaNCA_34806414967207
// MI455X (gfx1250) — matrix operand order repaired and checked by reading, not compiled, not run on hardware
//
#include <hip/hip_runtime.h>
#include <hip/hip_bf16.h>

typedef __attribute__((ext_vector_type(16))) __bf16 v16bf;
typedef __attribute__((ext_vector_type(8)))  float  v8f;

#define K_DIM 2048   // inner dim (weight rows / X cols)
#define M_DIM 2048   // weight cols
#define B_DIM 8192   // X rows
#define TILE_M 128
#define TILE_N 128
#define TILE_K 32
#define LDSW 34      // halfwords per padded LDS row (even -> 4B aligned pairs)

__device__ __forceinline__ unsigned short f32_to_bf16_rne(float f) {
    unsigned int u = __float_as_uint(f);
    unsigned int r = u + 0x7FFFu + ((u >> 16) & 1u);
    return (unsigned short)(r >> 16);
}
__device__ __forceinline__ float bf16_bits_to_f32(unsigned short h) {
    return __uint_as_float(((unsigned int)h) << 16);
}

// ---------------- Kernel 1: row sums of weight [2048] ----------------
__global__ __launch_bounds__(256) void rowsum_kernel(const float* __restrict__ W,
                                                     float* __restrict__ rowsum) {
    __shared__ float red[256];
    int row = blockIdx.x;
    int tid = threadIdx.x;
    float s = 0.f;
    #pragma unroll
    for (int e = 0; e < M_DIM / 256; ++e)
        s += W[row * M_DIM + e * 256 + tid];
    red[tid] = s;
    __syncthreads();
    for (int st = 128; st > 0; st >>= 1) {
        if (tid < st) red[tid] += red[tid + st];
        __syncthreads();
    }
    if (tid == 0) rowsum[row] = red[0];
}

// ---------------- Kernel 2: col sums of weight [2048] ----------------
__global__ __launch_bounds__(256) void colsum_kernel(const float* __restrict__ W,
                                                     float* __restrict__ colsum) {
    int col = blockIdx.x * 256 + threadIdx.x;
    float s = 0.f;
    for (int r = 0; r < K_DIM; ++r)
        s += W[r * M_DIM + col];
    colsum[col] = s;
}

// ---- Kernel 3: per-cell NCA MLP -> new_weight as bf16 hi/lo split ----
__global__ __launch_bounds__(256) void cell_mlp_kernel(
        const float* __restrict__ W,
        const float* __restrict__ rowsum, const float* __restrict__ colsum,
        const float* __restrict__ W1, const float* __restrict__ b1,
        const float* __restrict__ W2, const float* __restrict__ b2,
        const float* __restrict__ W3, const float* __restrict__ b3,
        unsigned short* __restrict__ wn_hi, unsigned short* __restrict__ wn_lo) {
    __shared__ float sW1[30], sb1[10], sW2[100], sb2[10], sW3[10], sb3[1];
    int tid = threadIdx.x;
    if (tid < 30)  sW1[tid] = W1[tid];
    if (tid < 100) sW2[tid] = W2[tid];
    if (tid < 10) { sb1[tid] = b1[tid]; sb2[tid] = b2[tid]; sW3[tid] = W3[tid]; }
    if (tid == 0)  sb3[0] = b3[0];
    __syncthreads();

    unsigned idx = blockIdx.x * 256 + tid;      // 0 .. 2048*2048-1
    int i = idx >> 11;                          // row
    int j = idx & (M_DIM - 1);                  // col
    float w  = W[idx];
    const float invn = 1.f / 2047.f;
    float f1 = (colsum[j] - w) * invn;          // leave-one-out col mean
    float f2 = (rowsum[i] - w) * invn;          // leave-one-out row mean

    float h1[10], h2[10];
    #pragma unroll
    for (int k = 0; k < 10; ++k) {
        float v = w * sW1[k] + f1 * sW1[10 + k] + f2 * sW1[20 + k] + sb1[k];
        h1[k] = v > 0.f ? v : 0.f;
    }
    #pragma unroll
    for (int k = 0; k < 10; ++k) {
        float v = sb2[k];
        #pragma unroll
        for (int c = 0; c < 10; ++c) v += h1[c] * sW2[c * 10 + k];
        h2[k] = v > 0.f ? v : 0.f;
    }
    float u = sb3[0];
    #pragma unroll
    for (int c = 0; c < 10; ++c) u += h2[c] * sW3[c];

    float nw = w + u;
    unsigned short hi = f32_to_bf16_rne(nw);
    float lo = nw - bf16_bits_to_f32(hi);
    wn_hi[idx] = hi;
    wn_lo[idx] = f32_to_bf16_rne(lo);
}

// ---- Kernel 4: logits = X @ new_weight via bf16x3 split WMMA ----
struct Frag {
    union { v16bf v; unsigned int u[8]; };
};

// registers holding the next K-slab while current one is computed
struct Prefetch {
    float xa[16];
    unsigned int bh[8], bl[8];   // packed pairs of bf16 (2 consecutive n)
};

// 32-bit base-offset addressing: per-thread bases computed once, the per-e
// deltas are compile-time immediates, and bases advance by a constant per
// K-stage.  X element e lives at  xoff + e*(8*K_DIM);  W pair e lives at
// woff + e*(4*M_DIM)  (elements, not bytes).
__device__ __forceinline__ void prefetch_global(
        const float* __restrict__ X,
        const unsigned short* __restrict__ Whi,
        const unsigned short* __restrict__ Wlo,
        unsigned& xoff, unsigned& woff, Prefetch& p) {
    #pragma unroll
    for (int e = 0; e < 16; ++e)
        p.xa[e] = X[xoff + (unsigned)(e * 8 * K_DIM)];
    #pragma unroll
    for (int e = 0; e < 8; ++e) {
        p.bh[e] = *(const unsigned int*)&Whi[woff + (unsigned)(e * 4 * M_DIM)];
        p.bl[e] = *(const unsigned int*)&Wlo[woff + (unsigned)(e * 4 * M_DIM)];
    }
    xoff += TILE_K;                 // next K-slab of X
    woff += TILE_K * M_DIM;         // next K-slab of W
}

__device__ __forceinline__ void stage_store(
        const Prefetch& p, unsigned aBase, unsigned bBase,
        unsigned short* aHi, unsigned short* aLo,
        unsigned short* bHi, unsigned short* bLo) {
    #pragma unroll
    for (int e = 0; e < 16; ++e) {
        unsigned a = aBase + (unsigned)(e * 8 * LDSW);
        float f = p.xa[e];
        unsigned short hi = f32_to_bf16_rne(f);
        aHi[a] = hi;
        aLo[a] = f32_to_bf16_rne(f - bf16_bits_to_f32(hi));
    }
    #pragma unroll
    for (int e = 0; e < 8; ++e) {
        unsigned b = bBase + (unsigned)(e * 4);
        bHi[b]        = (unsigned short)(p.bh[e] & 0xFFFFu);
        bHi[b + LDSW] = (unsigned short)(p.bh[e] >> 16);
        bLo[b]        = (unsigned short)(p.bl[e] & 0xFFFFu);
        bLo[b + LDSW] = (unsigned short)(p.bl[e] >> 16);
    }
}

__device__ __forceinline__ void stage_compute(
        const unsigned short* aHi, const unsigned short* aLo,
        const unsigned short* bHi, const unsigned short* bLo,
        int wm, int wn, int row16, int halfsel, v8f acc[2][4]) {
    // A fragments: 16-bit A 16x32 layout (ISA 7.12.2)
    Frag a_hi[2], a_lo[2];
    #pragma unroll
    for (int i = 0; i < 2; ++i) {
        unsigned mrow = (unsigned)(wm * 32 + i * 16 + row16) * LDSW;
        #pragma unroll
        for (int v = 0; v < 8; ++v) {
            unsigned kk = mrow + (unsigned)((v & 4) * 4 + (v & 3) * 2 + halfsel * 8);
            a_hi[i].u[v] = *(const unsigned int*)&aHi[kk];
            a_lo[i].u[v] = *(const unsigned int*)&aLo[kk];
        }
    }
    #pragma unroll
    for (int j = 0; j < 4; ++j) {
        unsigned ncol = (unsigned)(wn * 64 + j * 16 + row16) * LDSW;
        Frag b_hi, b_lo;
        #pragma unroll
        for (int v = 0; v < 8; ++v) {
            unsigned kk = ncol + (unsigned)((v & 4) * 4 + (v & 3) * 2 + halfsel * 8);
            b_hi.u[v] = *(const unsigned int*)&bHi[kk];
            b_lo.u[v] = *(const unsigned int*)&bLo[kk];
        }
        #pragma unroll
        for (int i = 0; i < 2; ++i) {
            acc[i][j] = __builtin_amdgcn_wmma_f32_16x16x32_bf16(
                false, a_hi[i].v, false, b_hi.v, (short)0, acc[i][j], false, false);
            acc[i][j] = __builtin_amdgcn_wmma_f32_16x16x32_bf16(
                false, a_hi[i].v, false, b_lo.v, (short)0, acc[i][j], false, false);
            acc[i][j] = __builtin_amdgcn_wmma_f32_16x16x32_bf16(
                false, a_lo[i].v, false, b_hi.v, (short)0, acc[i][j], false, false);
        }
    }
}

__global__ __launch_bounds__(256) void gemm_bf16x3_kernel(
        const float* __restrict__ X,
        const unsigned short* __restrict__ Whi,
        const unsigned short* __restrict__ Wlo,
        float* __restrict__ out) {
    // double-buffered LDS stages (ping-pong)
    __shared__ unsigned short A0h[TILE_M * LDSW], A0l[TILE_M * LDSW];
    __shared__ unsigned short B0h[TILE_N * LDSW], B0l[TILE_N * LDSW];
    __shared__ unsigned short A1h[TILE_M * LDSW], A1l[TILE_M * LDSW];
    __shared__ unsigned short B1h[TILE_N * LDSW], B1l[TILE_N * LDSW];

    const int tid  = threadIdx.x;
    const int lane = tid & 31;
    const int wid  = tid >> 5;            // 8 waves
    const int wm   = wid & 3;             // 4 M-groups of 32 rows
    const int wn   = wid >> 2;            // 2 N-groups of 64 cols
    const int bm   = blockIdx.y * TILE_M;
    const int bn   = blockIdx.x * TILE_N;

    const int row16   = lane & 15;
    const int halfsel = lane >> 4;        // lane group (ISA 16-bit frag layout)

    // per-thread 32-bit base offsets (elements)
    unsigned xoff  = (unsigned)(bm + (tid >> 5)) * K_DIM + (unsigned)(tid & 31);
    unsigned woff  = (unsigned)(tid >> 6) * M_DIM + (unsigned)bn + (unsigned)((tid & 63) * 2);
    const unsigned aBase = (unsigned)(tid >> 5) * LDSW + (unsigned)(tid & 31);
    const unsigned bBase = (unsigned)((tid & 63) * 2) * LDSW + (unsigned)(tid >> 6);

    v8f acc[2][4] = {};                   // 2x4 tiles of 16x16 f32

    Prefetch pf;
    prefetch_global(X, Whi, Wlo, xoff, woff, pf);
    stage_store(pf, aBase, bBase, A0h, A0l, B0h, B0l);
    __syncthreads();

    // 64 K-stages, manually unrolled 2x over the ping-pong buffers.
    for (int k0 = 0; k0 < K_DIM; k0 += 2 * TILE_K) {
        // stage s (buffer 0): prefetch s+1 first so global latency overlaps WMMA
        prefetch_global(X, Whi, Wlo, xoff, woff, pf);
        stage_compute(A0h, A0l, B0h, B0l, wm, wn, row16, halfsel, acc);
        stage_store(pf, aBase, bBase, A1h, A1l, B1h, B1l);
        __syncthreads();

        // stage s+1 (buffer 1): prefetch s+2 unless past the end
        if (k0 + 2 * TILE_K < K_DIM) {
            prefetch_global(X, Whi, Wlo, xoff, woff, pf);
            stage_compute(A1h, A1l, B1h, B1l, wm, wn, row16, halfsel, acc);
            stage_store(pf, aBase, bBase, A0h, A0l, B0h, B0l);
        } else {
            stage_compute(A1h, A1l, B1h, B1l, wm, wn, row16, halfsel, acc);
        }
        __syncthreads();
    }

    // epilogue: f32 C/D 16x16 layout -> global logits
    #pragma unroll
    for (int i = 0; i < 2; ++i) {
        #pragma unroll
        for (int j = 0; j < 4; ++j) {
            #pragma unroll
            for (int r = 0; r < 8; ++r) {
                unsigned row = (unsigned)(bm + wm * 32 + i * 16 + r + halfsel * 8);
                unsigned col = (unsigned)(bn + wn * 64 + j * 16 + row16);
                out[row * (unsigned)M_DIM + col] = acc[i][j][r];
            }
        }
    }
}

// ---------------- Kernel 5: in-place row softmax ----------------
__global__ __launch_bounds__(256) void softmax_kernel(float* __restrict__ out) {
    __shared__ float red[256];
    const unsigned row = blockIdx.x;
    const int tid = threadIdx.x;
    float v[M_DIM / 256];
    float lmax = -3.4e38f;
    #pragma unroll
    for (int e = 0; e < M_DIM / 256; ++e) {
        v[e] = out[row * M_DIM + e * 256 + tid];
        lmax = fmaxf(lmax, v[e]);
    }
    red[tid] = lmax;
    __syncthreads();
    for (int st = 128; st > 0; st >>= 1) {
        if (tid < st) red[tid] = fmaxf(red[tid], red[tid + st]);
        __syncthreads();
    }
    float m = red[0];
    __syncthreads();
    float ls = 0.f;
    #pragma unroll
    for (int e = 0; e < M_DIM / 256; ++e) {
        v[e] = expf(v[e] - m);
        ls += v[e];
    }
    red[tid] = ls;
    __syncthreads();
    for (int st = 128; st > 0; st >>= 1) {
        if (tid < st) red[tid] += red[tid + st];
        __syncthreads();
    }
    float inv = 1.f / red[0];
    #pragma unroll
    for (int e = 0; e < M_DIM / 256; ++e)
        out[row * M_DIM + e * 256 + tid] = v[e] * inv;
}

extern "C" void kernel_launch(void* const* d_in, const int* in_sizes, int n_in,
                              void* d_out, int out_size, void* d_ws, size_t ws_size,
                              hipStream_t stream) {
    const float* X  = (const float*)d_in[0];
    const float* W  = (const float*)d_in[1];
    const float* W1 = (const float*)d_in[2];
    const float* b1 = (const float*)d_in[3];
    const float* W2 = (const float*)d_in[4];
    const float* b2 = (const float*)d_in[5];
    const float* W3 = (const float*)d_in[6];
    const float* b3 = (const float*)d_in[7];
    float* out = (float*)d_out;

    char* ws = (char*)d_ws;
    const size_t nElems = (size_t)K_DIM * M_DIM;            // 4M
    unsigned short* wn_hi = (unsigned short*)ws;            // 8 MB
    unsigned short* wn_lo = (unsigned short*)(ws + nElems * 2);
    float* rowsum = (float*)(ws + nElems * 4);
    float* colsum = rowsum + K_DIM;

    rowsum_kernel<<<K_DIM, 256, 0, stream>>>(W, rowsum);
    colsum_kernel<<<M_DIM / 256, 256, 0, stream>>>(W, colsum);
    cell_mlp_kernel<<<(int)(nElems / 256), 256, 0, stream>>>(
        W, rowsum, colsum, W1, b1, W2, b2, W3, b3, wn_hi, wn_lo);

    dim3 grid(M_DIM / TILE_N, B_DIM / TILE_M);              // 16 x 64 blocks
    gemm_bf16x3_kernel<<<grid, 256, 0, stream>>>(X, wn_hi, wn_lo, out);

    softmax_kernel<<<B_DIM, 256, 0, stream>>>(out);
}
